// BondMatrixMessage_76647986364766
// MI455X (gfx1250) — matrix operand order repaired and checked by reading, not compiled, not run on hardware
//
#include <hip/hip_runtime.h>

typedef __bf16        v16bf __attribute__((ext_vector_type(16)));
typedef float         v8f   __attribute__((ext_vector_type(8)));
typedef float         v4f   __attribute__((ext_vector_type(4)));
typedef unsigned int  v8u   __attribute__((ext_vector_type(8)));

#define BATCH   8
#define NATOMS  4096
#define NEDGES  16384
#define ADIM    32
#define KDIM    64
#define ET_PER_B (NEDGES / 16)                 // edge-tiles per batch = 1024
#define TILES_TOTAL (BATCH * ET_PER_B)         // 8192
#define WLDS_DWORDS (64 * 2 * 8 * 32)          // 32768 dwords = 128 KB

__global__ __launch_bounds__(256) void zero_out_kernel(float* __restrict__ out, int n4) {
    int i = blockIdx.x * blockDim.x + threadIdx.x;
    int stride = gridDim.x * blockDim.x;
    v4f z = {0.f, 0.f, 0.f, 0.f};
    for (; i < n4; i += stride) ((v4f*)out)[i] = z;
}

__global__ __launch_bounds__(256) void bond_msg_kernel(
        const float* __restrict__ atom_state,   // (B, N, 32) f32
        const float* __restrict__ bond_state,   // (B, E, 64) f32
        const float* __restrict__ bond_transform, // (64, 1024) f32, [k][i*32+j]
        const int*   __restrict__ conn,         // (B, E, 2) i32
        float* __restrict__ out) {              // (B, N, 32) f32, pre-zeroed

    // ---- Stage 1: bond_transform -> LDS in WMMA B-operand bf16 layout ----
    // B chunk (c = j*64+k ordering): B[t, i] = W[k0 + t, i, j],  k0 = 32*(chunk&1), j = chunk>>1
    // 16-bit B layout: VGPR v, lane<16: K={2v,2v+1}(+8 if v>=4), N=lane ; lane>=16: K={8+2v,8+2v+1}(+8 if v>=4), N=lane-16
    // LDS index = ((chunk*2 + half)*8 + v)*32 + lane   (lane fastest -> conflict-free ds_load_b32)
    __shared__ unsigned int Wlds[WLDS_DWORDS];

    const int tid = threadIdx.x;
    for (int idx = tid; idx < WLDS_DWORDS; idx += 256) {
        const int lane  = idx & 31;
        const int v     = (idx >> 5) & 7;
        const int half  = (idx >> 8) & 1;
        const int chunk = idx >> 9;              // 0..63
        const int j     = chunk >> 1;
        const int kr    = chunk & 1;
        const int i     = half * 16 + (lane & 15);
        const int tb    = ((lane < 16) ? 0 : 8) + 2 * v + ((v >= 4) ? 8 : 0);
        const int k0    = kr * 32 + tb;
        union { __bf16 h[2]; unsigned int u; } pk;
        pk.h[0] = (__bf16)bond_transform[(k0    ) * 1024 + i * 32 + j];
        pk.h[1] = (__bf16)bond_transform[(k0 + 1) * 1024 + i * 32 + j];
        Wlds[idx] = pk.u;
    }
    __syncthreads();

    // ---- Stage 2: per-wave 16-edge tiles ----
    const int lane   = tid & 31;
    const int wave   = tid >> 5;
    const int waveG  = blockIdx.x * 8 + wave;
    const int nWaves = gridDim.x * 8;
    const int row    = lane & 15;                // edge row within tile
    const int hiHalf = (lane >= 16) ? 1 : 0;
    const int t0     = hiHalf ? 8 : 0;           // A-layout K-slot base for this lane

    for (int tile = waveG; tile < TILES_TOTAL; tile += nWaves) {
        const int b    = tile / ET_PER_B;
        const int e0   = (tile % ET_PER_B) * 16;
        const int edge = e0 + row;

        const int src = conn[((size_t)b * NEDGES + edge) * 2 + 0];

        // x row: atom_state[b, src, 0..31]  (f32 in regs; 8x b128 loads)
        const float* xr = atom_state + ((size_t)b * NATOMS + src) * ADIM;
        float xrow[32];
#pragma unroll
        for (int q = 0; q < 8; ++q) ((v4f*)xrow)[q] = ((const v4f*)xr)[q];

        // S operands: bond_state row in the 16-bit A-operand layout (bf16)
        //   lane<16 holds K-slots {0..7,16..23}; lane>=16 holds {8..15,24..31}
        const float* sr = bond_state + ((size_t)b * NEDGES + edge) * KDIM;
        v16bf S[2];
#pragma unroll
        for (int kr = 0; kr < 2; ++kr) {
#pragma unroll
            for (int q = 0; q < 8; ++q) {
                S[kr][q]     = (__bf16)sr[kr * 32 + t0 + q];
                S[kr][8 + q] = (__bf16)sr[kr * 32 + t0 + 16 + q];
            }
        }

        v8f acc0 = {};   // output cols 0..15
        v8f acc1 = {};   // output cols 16..31

#pragma unroll 4
        for (int j = 0; j < 32; ++j) {
            const __bf16 xb = (__bf16)xrow[j];
            v16bf xs;
#pragma unroll
            for (int q = 0; q < 16; ++q) xs[q] = xb;

#pragma unroll
            for (int kr = 0; kr < 2; ++kr) {
                // A tile = outer(s, x) slice: A[e, t] = s[e, k0+t] * x[e, j]
                v16bf A = xs * S[kr];            // packed bf16 multiply

                const int chunk = j * 2 + kr;
                v8u w0, w1;
#pragma unroll
                for (int v = 0; v < 8; ++v) {
                    w0[v] = Wlds[((chunk * 2 + 0) * 8 + v) * 32 + lane];
                    w1[v] = Wlds[((chunk * 2 + 1) * 8 + v) * 32 + lane];
                }
                v16bf B0 = __builtin_bit_cast(v16bf, w0);
                v16bf B1 = __builtin_bit_cast(v16bf, w1);

                acc0 = __builtin_amdgcn_wmma_f32_16x16x32_bf16(
                           false, A, false, B0, (short)0, acc0, false, false);
                acc1 = __builtin_amdgcn_wmma_f32_16x16x32_bf16(
                           false, A, false, B1, (short)0, acc1, false, false);
            }
        }

        // ---- Scatter: C/D layout row m = r + 8*hiHalf is edge e0+m, col = lane&15 ----
        const int col = lane & 15;
#pragma unroll
        for (int r = 0; r < 8; ++r) {
            const int m   = r + hiHalf * 8;
            const int tgt = conn[((size_t)b * NEDGES + e0 + m) * 2 + 1];
            float* op = out + ((size_t)b * NATOMS + tgt) * ADIM + col;
            unsafeAtomicAdd(op,      acc0[r]);
            unsafeAtomicAdd(op + 16, acc1[r]);
        }
    }
}

extern "C" void kernel_launch(void* const* d_in, const int* in_sizes, int n_in,
                              void* d_out, int out_size, void* d_ws, size_t ws_size,
                              hipStream_t stream) {
    const float* atom_state     = (const float*)d_in[0];
    const float* bond_state     = (const float*)d_in[1];
    const float* bond_transform = (const float*)d_in[2];
    const int*   conn           = (const int*)d_in[3];
    float* out = (float*)d_out;

    // out is poisoned by the harness and we accumulate atomically -> zero it first
    const int n4 = (BATCH * NATOMS * ADIM) / 4;
    zero_out_kernel<<<256, 256, 0, stream>>>(out, n4);

    bond_msg_kernel<<<256, 256, 0, stream>>>(atom_state, bond_state,
                                             bond_transform, conn, out);
}
